// QuantizedLinear_11278584119428
// MI455X (gfx1250) — compile-verified, matrix operand order checked by reading, not run on hardware
//
#include <hip/hip_runtime.h>
#include <hip/hip_bf16.h>

// ---------------------------------------------------------------------------
// NVFP4 quantized linear for MI455X (gfx1250, wave32).
//   out[256,13824] = (a_fp4*s_a) @ (w_fp4*s_w)^T * alpha + bias,  K=5120
// HBM-bound on the single 283 MB f32-held-fp4 weight read. BM=256 => weights
// read exactly once; dequant on the fly to bf16 (exact: fp4*e4m3 <= 6
// mantissa bits) into LDS and feed v_wmma_f32_16x16x32_bf16.
// A-tile uses GLOBAL_LOAD_ASYNC_TO_LDS_B128 (ASYNCcnt) when available.
// ---------------------------------------------------------------------------

typedef __attribute__((ext_vector_type(16))) __bf16 v16bf;
typedef __attribute__((ext_vector_type(8)))  float  v8f;
typedef __attribute__((ext_vector_type(4)))  float  v4f;
typedef __attribute__((ext_vector_type(4)))  int    v4i;

typedef __attribute__((address_space(1))) v4i* gptr_v4i;  // global
typedef __attribute__((address_space(3))) v4i* lptr_v4i;  // LDS

#define M_DIM 256
#define N_DIM 13824
#define K_DIM 5120
#define BN    64
#define BK    32
#define SROW  40   // 32 bf16 + 8 pad -> 80B row stride, conflict-free b128 reads

#if defined(__has_builtin)
#  if __has_builtin(__builtin_amdgcn_global_load_async_to_lds_b128)
#    define HAS_ASYNC_LDS 1
#  else
#    define HAS_ASYNC_LDS 0
#  endif
#  if __has_builtin(__builtin_amdgcn_s_wait_asynccnt)
#    define HAS_WAIT_ASYNC 1
#  else
#    define HAS_WAIT_ASYNC 0
#  endif
#  if __has_builtin(__builtin_amdgcn_cvt_pk_bf16_f32)
#    define HAS_CVT_PK_BF16 1
#  else
#    define HAS_CVT_PK_BF16 0
#  endif
#else
#  define HAS_ASYNC_LDS 0
#  define HAS_WAIT_ASYNC 0
#  define HAS_CVT_PK_BF16 0
#endif

// round f32 -> bf16 (RNE), return as f32 value
__device__ __forceinline__ float bf16r(float f) {
    unsigned u = __float_as_uint(f);
    u = (u + 0x7FFFu + ((u >> 16) & 1u)) & 0xFFFF0000u;
    return __uint_as_float(u);
}
// f32 -> bf16 bits (RNE)
__device__ __forceinline__ unsigned short f2bf(float f) {
    unsigned u = __float_as_uint(f);
    return (unsigned short)((u + 0x7FFFu + ((u >> 16) & 1u)) >> 16);
}
// pack two f32 -> two bf16 (RNE); uses v_cvt_pk_bf16_f32 when available
__device__ __forceinline__ unsigned pk2bf(float a, float b) {
#if HAS_CVT_PK_BF16
    auto v = __builtin_amdgcn_cvt_pk_bf16_f32(a, b);
    return __builtin_bit_cast(unsigned, v);
#else
    return (unsigned)f2bf(a) | ((unsigned)f2bf(b) << 16);
#endif
}
// round non-negative v (<=448) to fp8 e4m3 grid, RNE
__device__ __forceinline__ float e4m3_round(float v) {
    if (v <= 0.0f) return 0.0f;
    if (v < 0.015625f) {                       // e4m3 subnormal range (<2^-6)
        return rintf(v * 512.0f) * (1.0f / 512.0f);
    }
    unsigned b = __float_as_uint(v);
    unsigned lsb = (b >> 20) & 1u;
    b += 0x7FFFFu + lsb;                       // RNE to 3 mantissa bits
    b &= 0xFFF00000u;
    float r = __uint_as_float(b);
    return r > 448.0f ? 448.0f : r;
}
// round to nearest fp4 e2m1 value, ties away from zero (matches searchsorted 'right')
__device__ __forceinline__ float fp4_round(float v) {
    float a = fabsf(v);
    float r;
    if      (a < 0.25f) r = 0.0f;
    else if (a < 0.75f) r = 0.5f;
    else if (a < 1.25f) r = 1.0f;
    else if (a < 1.75f) r = 1.5f;
    else if (a < 2.5f)  r = 2.0f;
    else if (a < 3.5f)  r = 3.0f;
    else if (a < 5.0f)  r = 4.0f;
    else                r = 6.0f;
    return copysignf(r, v);
}

// ---- kernel 1a: init amax ----
__global__ void k_init(unsigned* amax_bits) { *amax_bits = 0u; }

// ---- kernel 1b: abs-max of bf16-rounded activations ----
__global__ void k_amax(const float* __restrict__ x, unsigned* amax_bits, int n) {
    float m = 0.0f;
    for (int i = blockIdx.x * blockDim.x + threadIdx.x; i < n;
         i += gridDim.x * blockDim.x) {
        m = fmaxf(m, fabsf(bf16r(x[i])));
    }
    __shared__ float red[256];
    red[threadIdx.x] = m;
    __syncthreads();
    for (int s = 128; s > 0; s >>= 1) {
        if ((int)threadIdx.x < s) red[threadIdx.x] = fmaxf(red[threadIdx.x], red[threadIdx.x + s]);
        __syncthreads();
    }
    if (threadIdx.x == 0) atomicMax(amax_bits, __float_as_uint(red[0]));
}

// ---- kernel 2: quantize activations -> dequantized bf16 a_u in workspace ----
// one thread per 16-element block: M*K/16 = 81920 threads
__global__ void k_actquant(const float* __restrict__ x,
                           const unsigned* __restrict__ amax_bits,
                           unsigned short* __restrict__ a_u) {
    int idx = blockIdx.x * blockDim.x + threadIdx.x;     // 0..81919
    int row = idx / (K_DIM / 16);
    int blk = idx - row * (K_DIM / 16);
    const float* src = x + (size_t)row * K_DIM + blk * 16;

    float v[16], am = 0.0f;
#pragma unroll
    for (int i = 0; i < 16; ++i) { v[i] = bf16r(src[i]); am = fmaxf(am, fabsf(v[i])); }

    float amax = __uint_as_float(*amax_bits);
    float a_gs = 2688.0f / fmaxf(amax, 1e-20f);          // (448*6)/amax
    float sc   = e4m3_round(fminf(am * (a_gs * (1.0f / 6.0f)), 448.0f));
    float safe = (sc == 0.0f) ? 1.0f : sc;
    float inv  = a_gs / safe;

    unsigned short* dst = a_u + (size_t)row * K_DIM + blk * 16;
#pragma unroll
    for (int i = 0; i < 16; ++i) dst[i] = f2bf(fp4_round(v[i] * inv) * sc);
}

// ---- kernel 3: block-tiled bf16 WMMA GEMM with on-the-fly weight dequant ----
// grid = N/64 blocks, 256 threads (8 waves). Tile: 256(M) x 64(N), K-step 32.
// Each wave: 4 M-tiles x 2 N-tiles of 16x16 (8 WMMAs / K-step).
__global__ void __launch_bounds__(256)
k_gemm(const unsigned short* __restrict__ a_u,   // [256,5120] bf16 (dequantized)
       const float* __restrict__ W,              // [13824,5120] fp4 values in f32
       const float* __restrict__ wscale,         // [13824,320]
       const float* __restrict__ w_gs,           // scalar
       const float* __restrict__ bias,           // [13824]
       const unsigned* __restrict__ amax_bits,
       float* __restrict__ out) {
    __shared__ unsigned short sA[M_DIM][SROW];   // 20.0 KB
    __shared__ unsigned short sB[BN][SROW];      //  5.0 KB

    const int tid  = threadIdx.x;
    const int lane = tid & 31;
    const int wv   = tid >> 5;        // 0..7
    const int mg   = wv & 3;          // M group: rows mg*64 .. +63
    const int ng   = wv >> 2;         // N group: cols ng*32 .. +31
    const int lh   = lane & 15;       // row/col within 16x16 tile
    const int kb   = (lane >> 4) * 8; // K half select (ISA 16-bit A/B layout)
    const int n0   = blockIdx.x * BN;

    const v8f zero = {0, 0, 0, 0, 0, 0, 0, 0};
    v8f acc[4][2];
#pragma unroll
    for (int mt = 0; mt < 4; ++mt)
#pragma unroll
        for (int nt = 0; nt < 2; ++nt) acc[mt][nt] = zero;

    // B-load mapping: 512 float4 chunks (64 rows x 8), 2 per thread
    const int chunk0 = tid;            // + t*256
    for (int k0 = 0; k0 < K_DIM; k0 += BK) {
        __syncthreads();  // previous iteration finished reading LDS

        // A: row = tid, copy 32 bf16 global -> LDS (async path uses ASYNCcnt
        // and writes LDS directly, no VGPR round-trip)
#if HAS_ASYNC_LDS
        {
            gptr_v4i gsrc = (gptr_v4i)(a_u + (size_t)tid * K_DIM + k0);
            lptr_v4i ldst = (lptr_v4i)&sA[tid][0];
#pragma unroll
            for (int c = 0; c < 4; ++c)
                __builtin_amdgcn_global_load_async_to_lds_b128(gsrc + c, ldst + c, 0, 0);
        }
#else
        {
            const v4f* src = (const v4f*)(a_u + (size_t)tid * K_DIM + k0);
            v4f* dst = (v4f*)&sA[tid][0];
#pragma unroll
            for (int c = 0; c < 4; ++c) dst[c] = src[c];
        }
#endif
        // B: dequantize weights f32 -> bf16 into LDS
#pragma unroll
        for (int t = 0; t < 2; ++t) {
            int chunk = chunk0 + t * 256;          // 0..511
            int row = chunk >> 3;                  // 0..63
            int kc  = (chunk & 7) * 4;             // 0,4,..,28
            const float4 wvv = *(const float4*)(W + (size_t)(n0 + row) * K_DIM + k0 + kc);
            float sc = wscale[(size_t)(n0 + row) * (K_DIM / 16) + ((k0 + kc) >> 4)];
            uint2 pk;
            pk.x = pk2bf(wvv.x * sc, wvv.y * sc);
            pk.y = pk2bf(wvv.z * sc, wvv.w * sc);
            *(uint2*)&sB[row][kc] = pk;
        }
        // prefetch next K-step of the weight stream (global_prefetch_b8)
        if (k0 + BK < K_DIM) {
            int row = chunk0 >> 3, kc = (chunk0 & 7) * 4;
            __builtin_prefetch(W + (size_t)(n0 + row) * K_DIM + k0 + BK + kc, 0, 0);
        }
#if HAS_ASYNC_LDS
#  if HAS_WAIT_ASYNC
        __builtin_amdgcn_s_wait_asynccnt(0);
#  else
        asm volatile("s_wait_asynccnt 0x0" ::: "memory");
#  endif
#endif
        __syncthreads();

        // fragments (ds_read_b128 pairs, layout per ISA 7.12.2 16-bit A/B)
        union Frag { v4f f[2]; v16bf v; };
        Frag af[4], bfv[2];
#pragma unroll
        for (int mt = 0; mt < 4; ++mt) {
            int r = mg * 64 + mt * 16 + lh;
            af[mt].f[0] = *(const v4f*)&sA[r][kb];
            af[mt].f[1] = *(const v4f*)&sA[r][kb + 16];
        }
#pragma unroll
        for (int nt = 0; nt < 2; ++nt) {
            int c = ng * 32 + nt * 16 + lh;
            bfv[nt].f[0] = *(const v4f*)&sB[c][kb];
            bfv[nt].f[1] = *(const v4f*)&sB[c][kb + 16];
        }
#pragma unroll
        for (int mt = 0; mt < 4; ++mt)
#pragma unroll
            for (int nt = 0; nt < 2; ++nt)
                acc[mt][nt] = __builtin_amdgcn_wmma_f32_16x16x32_bf16(
                    false, af[mt].v, false, bfv[nt].v,
                    (short)0, acc[mt][nt], false, false);
    }

    // epilogue: * alpha + bias
    float amax  = __uint_as_float(*amax_bits);
    float a_gs  = 2688.0f / fmaxf(amax, 1e-20f);
    float alpha = 1.0f / (a_gs * w_gs[0]);
#pragma unroll
    for (int mt = 0; mt < 4; ++mt) {
#pragma unroll
        for (int nt = 0; nt < 2; ++nt) {
            int n = n0 + ng * 32 + nt * 16 + lh;
            float b = bias[n];
#pragma unroll
            for (int j = 0; j < 8; ++j) {
                int m = mg * 64 + mt * 16 + (lane >> 4) * 8 + j;
                out[(size_t)m * N_DIM + n] = acc[mt][nt][j] * alpha + b;
            }
        }
    }
}

extern "C" void kernel_launch(void* const* d_in, const int* in_sizes, int n_in,
                              void* d_out, int out_size, void* d_ws, size_t ws_size,
                              hipStream_t stream) {
    const float* x      = (const float*)d_in[0];  // [256,5120]
    const float* w_fp4  = (const float*)d_in[1];  // [13824,5120]
    const float* wscale = (const float*)d_in[2];  // [13824,320]
    const float* w_gs   = (const float*)d_in[3];  // scalar
    const float* bias   = (const float*)d_in[4];  // [13824]
    float* out = (float*)d_out;

    // workspace: [0..255] amax bits, then bf16 a_u (256*5120*2 B = 2.62 MB)
    unsigned* amax_bits  = (unsigned*)d_ws;
    unsigned short* a_u  = (unsigned short*)((char*)d_ws + 256);

    k_init<<<1, 1, 0, stream>>>(amax_bits);
    k_amax<<<256, 256, 0, stream>>>(x, amax_bits, M_DIM * K_DIM);
    k_actquant<<<(M_DIM * (K_DIM / 16)) / 256, 256, 0, stream>>>(x, amax_bits, a_u);
    k_gemm<<<N_DIM / BN, 256, 0, stream>>>(a_u, w_fp4, wscale, w_gs, bias,
                                           amax_bits, out);
}
